// LFASR_2_to_8_extra_net_20117626814907
// MI455X (gfx1250) — matrix operand order repaired and checked by reading, not compiled, not run on hardware
//
#include <hip/hip_runtime.h>

typedef _Float16 h8  __attribute__((ext_vector_type(8)));
typedef _Float16 h16 __attribute__((ext_vector_type(16)));
typedef float    f8  __attribute__((ext_vector_type(8)));

#define HW 16384

union AB { h16 v; h8 p[2]; };

__device__ __forceinline__ float lrelu_f(float t) { return t < 0.f ? 0.2f * t : t; }

// ---------------------------------------------------------------------------
// View mapping for the 8x8 reconstruction grid (NEW_IND logic, hardcoded).
// Input views sit at grid (1,1),(1,6),(6,1),(6,6) -> lf_input channels 0..3.
// nv is planar f32 [b][64(60 used)][HW].
// ---------------------------------------------------------------------------
__device__ __forceinline__ float view_val(const float* __restrict__ lf,
                                          const float* __restrict__ nv,
                                          int b, int vidx, int pix) {
  const int r = vidx >> 3, c = vidx & 7;
  const bool is_in = ((r == 1) || (r == 6)) && ((c == 1) || (c == 6));
  if (is_in) {
    const int idx = ((r == 6) ? 2 : 0) + ((c == 6) ? 1 : 0);
    return lf[((unsigned long)b * 4 + idx) * HW + pix];
  }
  const int nvi = vidx - (vidx > 9) - (vidx > 14) - (vidx > 49) - (vidx > 54);
  return nv[((unsigned long)b * 64 + nvi) * HW + pix];
}

// ---------------------------------------------------------------------------
// 3x3 pad-1 conv on NHWC f16 activations, implicit GEMM via WMMA.
// Activations: [img][y][x][C] with 64-channel blocks; for CIN=256 the four
// 64-ch blocks live at img-relative stride CBS = 64*HW (the 4 angular views).
// Block = 256 thr (8 waves), computes 64 couts x one row (128 px).
// grid.x = IMGS*128.  Wave = 2 M-tiles x 2 N-tiles.
// Weights: Wg[64][K] f16, K = tap*CIN + cin (tap = ky*3+kx).
// ---------------------------------------------------------------------------
template <int CIN>
__global__ void __launch_bounds__(256)
conv3x3_wmma(const _Float16* __restrict__ in, unsigned long inStride,
             const _Float16* __restrict__ Wg,
             const float* __restrict__ bias, int coutValid, int doLrelu,
             _Float16* __restrict__ out16,
             float* __restrict__ out32, const float* __restrict__ res,
             unsigned long out32Stride) {
  extern __shared__ _Float16 lds[];     // [3][132][CIN], halos zeroed
  constexpr int CIN8 = CIN >> 3;
  constexpr int K = CIN * 9;
  constexpr int NCHUNK = K >> 5;
  const int tid = threadIdx.x;
  const int img = blockIdx.x >> 7;
  const int y   = blockIdx.x & 127;
  const _Float16* inImg = in + (unsigned long)img * inStride;
  const unsigned long CBS = 64UL * HW;

  // ---- stage NHWC tile: rows y-1..y+1, cols -1..130, all CIN channels ----
  constexpr int NSLOT = 3 * 132 * CIN8;
  for (int s = tid; s < NSLOT; s += 256) {
    const int cc = s & (CIN8 - 1);       // 8-channel chunk
    const int t  = s / CIN8;
    const int xm = t % 132;              // x = xm-1
    const int r  = t / 132;
    const int yy = y + r - 1;
    const int x  = xm - 1;
    h8 v;
    if (((unsigned)yy < 128u) && ((unsigned)x < 128u)) {
      v = *(const h8*)(inImg + (unsigned long)(cc >> 3) * CBS +
                       (unsigned long)(yy * 128 + x) * 64 + (cc & 7) * 8);
    } else {
#pragma unroll
      for (int e = 0; e < 8; ++e) v[e] = (_Float16)0.f;
    }
    *(h8*)(&lds[(unsigned long)(r * 132 + xm) * CIN + cc * 8]) = v;
  }
  __syncthreads();

  const int wave = tid >> 5;
  const int lane = tid & 31;
  const int mh  = wave & 1;    // M half (32 rows)
  const int nq  = wave >> 1;   // N quarter (32 cols)
  const int lhi = lane >> 4;
  const int llo = lane & 15;

  f8 acc[2][2];
#pragma unroll
  for (int i = 0; i < 2; ++i)
#pragma unroll
    for (int j = 0; j < 2; ++j)
#pragma unroll
      for (int e = 0; e < 8; ++e) acc[i][j][e] = 0.f;

  for (int ch = 0; ch < NCHUNK; ++ch) {
    const int k0  = ch << 5;
    const int tap = k0 / CIN;            // chunk lies within one (ky,kx) tap
    const int c0  = k0 - tap * CIN;
    const int ky  = tap / 3;
    const int kx  = tap - ky * 3;

    AB a[2];
#pragma unroll
    for (int mi = 0; mi < 2; ++mi) {
      const int mrow = mh * 32 + mi * 16 + llo;
      const _Float16* ap = Wg + (unsigned long)mrow * K + k0 + lhi * 8;
      a[mi].p[0] = *(const h8*)ap;         // K: 8*hi + 0..7
      a[mi].p[1] = *(const h8*)(ap + 16);  // K: 16 + 8*hi + 0..7
    }

    AB bf[2];
#pragma unroll
    for (int jn = 0; jn < 2; ++jn) {
      const int n = nq * 32 + jn * 16 + llo;
      const _Float16* bp =
          &lds[(unsigned long)(ky * 132 + n + kx) * CIN + c0 + lhi * 8];
      bf[jn].p[0] = *(const h8*)bp;        // K: 8*hi + 0..7
      bf[jn].p[1] = *(const h8*)(bp + 16); // K: 16 + 8*hi + 0..7
    }

#pragma unroll
    for (int mi = 0; mi < 2; ++mi)
#pragma unroll
      for (int jn = 0; jn < 2; ++jn)
        acc[mi][jn] = __builtin_amdgcn_wmma_f32_16x16x32_f16(
            false, a[mi].v, false, bf[jn].v, (short)0, acc[mi][jn],
            false, false);
  }

  const int yoff = y * 128;
#pragma unroll
  for (int mi = 0; mi < 2; ++mi) {
    const int m8 = mh * 32 + mi * 16 + lhi * 8;   // C/D: M = vgpr + 8*hi
#pragma unroll
    for (int jn = 0; jn < 2; ++jn) {
      const int n = nq * 32 + jn * 16 + llo;
      float tv[8];
#pragma unroll
      for (int rr = 0; rr < 8; ++rr) {
        float t = acc[mi][jn][rr] +
                  ((m8 + rr < coutValid) ? bias[m8 + rr] : 0.f);
        if (doLrelu) t = lrelu_f(t);
        tv[rr] = t;
      }
      if (out16) {   // NHWC f16, always 64 couts
        h8 ov;
#pragma unroll
        for (int rr = 0; rr < 8; ++rr) ov[rr] = (_Float16)tv[rr];
        *(h8*)(out16 + (unsigned long)img * CBS +
               (unsigned long)(yoff + n) * 64 + m8) = ov;
      }
      if (out32) {   // planar f32 (+ optional residual), only valid couts
#pragma unroll
        for (int rr = 0; rr < 8; ++rr) {
          const int m = m8 + rr;
          if (m < coutValid) {
            float t = tv[rr];
            const unsigned long o =
                (unsigned long)img * out32Stride + (unsigned long)m * HW +
                yoff + n;
            if (res) t += res[o];
            out32[o] = t;
          }
        }
      }
    }
  }
}

// ---------------------------------------------------------------------------
// Angular mixing GEMM per batch: Y[256][HW] = Wa[256][256] * X[256][HW],
// X/Y in NHWC [b*4+uv][pix][64].  No LDS: B fragments straight from global
// (K = channels contiguous).  bias per cout (m&63), LeakyReLU, f16 out.
// grid.x = B * 4 * 128.
// ---------------------------------------------------------------------------
__global__ void __launch_bounds__(256)
gemm256_wmma(const _Float16* __restrict__ X, const _Float16* __restrict__ Wa,
             const float* __restrict__ bias, _Float16* __restrict__ Y) {
  const int tid = threadIdx.x;
  const int b   = blockIdx.x >> 9;
  const int r   = blockIdx.x & 511;
  const int m0  = (r >> 7) << 6;
  const int n0c = (r & 127) << 7;
  const unsigned long bb = (unsigned long)b * 4 * HW * 64;

  const int wave = tid >> 5, lane = tid & 31;
  const int mh = wave & 1, nq = wave >> 1;
  const int lhi = lane >> 4, llo = lane & 15;

  f8 acc[2][2];
#pragma unroll
  for (int i = 0; i < 2; ++i)
#pragma unroll
    for (int j = 0; j < 2; ++j)
#pragma unroll
      for (int e = 0; e < 8; ++e) acc[i][j][e] = 0.f;

  for (int ch = 0; ch < 8; ++ch) {
    const int k0 = ch << 5;
    const int uv = k0 >> 6;       // source angular view
    const int c064 = k0 & 63;

    AB a[2];
#pragma unroll
    for (int mi = 0; mi < 2; ++mi) {
      const int mrow = m0 + mh * 32 + mi * 16 + llo;
      const _Float16* ap = Wa + (unsigned long)mrow * 256 + k0 + lhi * 8;
      a[mi].p[0] = *(const h8*)ap;
      a[mi].p[1] = *(const h8*)(ap + 16);
    }

    AB bf[2];
#pragma unroll
    for (int jn = 0; jn < 2; ++jn) {
      const int n = n0c + nq * 32 + jn * 16 + llo;
      const _Float16* bp = X + bb + (unsigned long)uv * HW * 64 +
                           (unsigned long)n * 64 + c064 + lhi * 8;
      bf[jn].p[0] = *(const h8*)bp;
      bf[jn].p[1] = *(const h8*)(bp + 16);
    }

#pragma unroll
    for (int mi = 0; mi < 2; ++mi)
#pragma unroll
      for (int jn = 0; jn < 2; ++jn)
        acc[mi][jn] = __builtin_amdgcn_wmma_f32_16x16x32_f16(
            false, a[mi].v, false, bf[jn].v, (short)0, acc[mi][jn],
            false, false);
  }

#pragma unroll
  for (int mi = 0; mi < 2; ++mi) {
    const int m8 = m0 + mh * 32 + mi * 16 + lhi * 8;  // 8 consecutive couts
#pragma unroll
    for (int jn = 0; jn < 2; ++jn) {
      const int n = n0c + nq * 32 + jn * 16 + llo;
      h8 ov;
#pragma unroll
      for (int rr = 0; rr < 8; ++rr)
        ov[rr] = (_Float16)lrelu_f(acc[mi][jn][rr] + bias[(m8 + rr) & 63]);
      *(h8*)(Y + bb + ((unsigned long)(m8 >> 6) * HW + n) * 64 + (m8 & 63)) =
          ov;
    }
  }
}

// ---------------------------------------------------------------------------
// First layer: folded angular taps (K<=36) -> direct VALU kernel.
// out NHWC: featA[img][pix][64], bias+lrelu.  grid.x = 16*HW/256.
// ---------------------------------------------------------------------------
__global__ void __launch_bounds__(256)
feat_conv(const float* __restrict__ lf, const float* __restrict__ w_feat,
          const float* __restrict__ b_feat, _Float16* __restrict__ out) {
  __shared__ float wl[64 * 81];
  for (int i = threadIdx.x; i < 64 * 81; i += 256) wl[i] = w_feat[i];
  __syncthreads();
  const int gid = blockIdx.x * 256 + threadIdx.x;
  const int img = gid >> 14;
  const int pix = gid & (HW - 1);
  const int b = img >> 2, u = (img >> 1) & 1, v = img & 1;
  const int y = pix >> 7, x = pix & 127;

  float acc[64];
#pragma unroll
  for (int co = 0; co < 64; ++co) acc[co] = b_feat[co];

  for (int i = 0; i < 3; ++i) {
    const int uu = u + i - 1;
    if ((unsigned)uu >= 2u) continue;
    for (int j = 0; j < 3; ++j) {
      const int vv = v + j - 1;
      if ((unsigned)vv >= 2u) continue;
      const float* src = lf + (unsigned long)(b * 4 + uu * 2 + vv) * HW;
      for (int ky = 0; ky < 3; ++ky) {
        const int yy = y + ky - 1;
        if ((unsigned)yy >= 128u) continue;
        for (int kx = 0; kx < 3; ++kx) {
          const int xx = x + kx - 1;
          if ((unsigned)xx >= 128u) continue;
          const float val  = src[yy * 128 + xx];
          const int   widx = (i * 3 + j) * 9 + ky * 3 + kx;
#pragma unroll
          for (int co = 0; co < 64; ++co) acc[co] += wl[co * 81 + widx] * val;
        }
      }
    }
  }
  _Float16* o = out + ((unsigned long)img * HW + pix) * 64;
#pragma unroll
  for (int g = 0; g < 8; ++g) {
    h8 ov;
#pragma unroll
    for (int e = 0; e < 8; ++e) ov[e] = (_Float16)lrelu_f(acc[g * 8 + e]);
    *(h8*)(o + g * 8) = ov;
  }
}

// ---------------------------------------------------------------------------
// r1: stride-2 angular fold (K=36) -> direct VALU kernel.  NHWC blocked
// output [(b*4+uv)][pix][ (du,dv,cin16)=64 ] so r2/rp are standard convs.
// grid.x = 4*16*HW/256.
// ---------------------------------------------------------------------------
__global__ void __launch_bounds__(256)
r1_conv(const float* __restrict__ lf, const float* __restrict__ nv,
        const float* __restrict__ w_r1, const float* __restrict__ b_r1,
        _Float16* __restrict__ r1buf) {
  __shared__ float wl[576];
  for (int i = threadIdx.x; i < 576; i += 256) wl[i] = w_r1[i];
  __syncthreads();
  const int gid = blockIdx.x * 256 + threadIdx.x;
  const int pix = gid & (HW - 1);
  const int t   = gid >> 14;
  const int b   = t >> 4;
  const int ap  = t & 15;
  const int a1 = ap >> 2, a2 = ap & 3;
  const int y = pix >> 7, x = pix & 127;

  float acc[16];
#pragma unroll
  for (int c = 0; c < 16; ++c) acc[c] = b_r1[c];

#pragma unroll
  for (int du = 0; du < 2; ++du)
#pragma unroll
    for (int dv = 0; dv < 2; ++dv) {
      const int vidx = (2 * a1 + du) * 8 + (2 * a2 + dv);
      for (int ky = 0; ky < 3; ++ky) {
        const int yy = y + ky - 1;
        if ((unsigned)yy >= 128u) continue;
        for (int kx = 0; kx < 3; ++kx) {
          const int xx = x + kx - 1;
          if ((unsigned)xx >= 128u) continue;
          const float val = view_val(lf, nv, b, vidx, yy * 128 + xx);
          const int wbase = (du * 2 + dv) * 9 + ky * 3 + kx;
#pragma unroll
          for (int c = 0; c < 16; ++c) acc[c] += wl[c * 36 + wbase] * val;
        }
      }
    }
  const int uv = (a1 >> 1) * 2 + (a2 >> 1);
  const int cb = ((a1 & 1) * 2 + (a2 & 1)) * 16;
  _Float16* o =
      r1buf + (((unsigned long)b * 4 + uv) * HW + pix) * 64 + cb;
#pragma unroll
  for (int g = 0; g < 2; ++g) {
    h8 ov;
#pragma unroll
    for (int e = 0; e < 8; ++e) ov[e] = (_Float16)lrelu_f(acc[g * 8 + e]);
    *(h8*)(o + g * 8) = ov;
  }
}

// ---------------------------------------------------------------------------
// Compose an output half [4][64][HW] f32 from new-views + input views.
// ---------------------------------------------------------------------------
__global__ void __launch_bounds__(256)
compose_views(const float* __restrict__ lf, const float* __restrict__ nv,
              float* __restrict__ out) {
  const int gid = blockIdx.x * 256 + threadIdx.x;
  const int pix = gid & (HW - 1);
  const int t = gid >> 14;
  const int b = t >> 6, v = t & 63;
  out[gid] = view_val(lf, nv, b, v, pix);
}

// --------------------------- weight repack kernels -------------------------
__global__ void __launch_bounds__(256)
repack_spa(const float* __restrict__ w, _Float16* __restrict__ Wg, int total) {
  const int gid = blockIdx.x * 256 + threadIdx.x;
  if (gid >= total) return;
  const int k = gid % 576;
  const int t = gid / 576;
  const int m = t & 63;
  const int l = t >> 6;
  const int tap = k >> 6, cin = k & 63;
  Wg[gid] = (_Float16)w[(((unsigned long)l * 64 + m) * 64 + cin) * 9 + tap];
}

__global__ void __launch_bounds__(256)
repack_ang(const float* __restrict__ w, _Float16* __restrict__ Wa) {
  const int gid = blockIdx.x * 256 + threadIdx.x;
  if (gid >= 6 * 256 * 256) return;
  const int k = gid & 255;
  const int t = gid >> 8;
  const int m = t & 255;
  const int l = t >> 8;
  const int uv = m >> 6, cout = m & 63;
  const int u = uv >> 1, v = uv & 1;
  const int upv = k >> 6, cin = k & 63;
  const int up = upv >> 1, vp = upv & 1;
  const int ky = 1 + up - u, kx = 1 + vp - v;   // always in [0,3)
  Wa[gid] = (_Float16)w[(((unsigned long)l * 64 + cout) * 64 + cin) * 9 +
                        ky * 3 + kx];
}

__global__ void __launch_bounds__(256)
repack_synrp(const float* __restrict__ w, _Float16* __restrict__ Wg,
             int coutValid) {
  const int gid = blockIdx.x * 256 + threadIdx.x;
  if (gid >= 64 * 2304) return;
  const int k = gid % 2304;
  const int m = gid / 2304;
  if (m >= coutValid) { Wg[gid] = (_Float16)0.f; return; }
  const int tap = k / 256;
  const int rem = k & 255;
  const int uv = rem >> 6, cin = rem & 63;
  const int u = uv >> 1, v = uv & 1;
  Wg[gid] =
      (_Float16)w[((((unsigned long)m * 64 + cin) * 2 + u) * 2 + v) * 9 + tap];
}

__global__ void __launch_bounds__(256)
repack_r2(const float* __restrict__ w, _Float16* __restrict__ Wg) {
  const int gid = blockIdx.x * 256 + threadIdx.x;
  if (gid >= 64 * 576) return;
  const int k = gid % 576;
  const int m = gid / 576;
  const int tap = k >> 6;
  const int cc  = k & 63;
  const int dudv = cc >> 4, cin = cc & 15;
  const int du = dudv >> 1, dv = dudv & 1;
  Wg[gid] = (_Float16)w[((((unsigned long)m * 16 + cin) * 2 + du) * 2 + dv) *
                            9 + tap];
}

// ---------------------------------------------------------------------------
extern "C" void kernel_launch(void* const* d_in, const int* in_sizes, int n_in,
                              void* d_out, int out_size, void* d_ws,
                              size_t ws_size, hipStream_t stream) {
  (void)in_sizes; (void)n_in; (void)out_size; (void)ws_size;
  const float* lf     = (const float*)d_in[0];
  const float* w_feat = (const float*)d_in[1];
  const float* b_feat = (const float*)d_in[2];
  const float* w_spa  = (const float*)d_in[3];
  const float* b_spa  = (const float*)d_in[4];
  const float* w_ang  = (const float*)d_in[5];
  const float* b_ang  = (const float*)d_in[6];
  const float* w_syn  = (const float*)d_in[7];
  const float* b_syn  = (const float*)d_in[8];
  const float* w_r1   = (const float*)d_in[9];
  const float* b_r1   = (const float*)d_in[10];
  const float* w_r2   = (const float*)d_in[11];
  const float* b_r2   = (const float*)d_in[12];
  const float* w_rp   = (const float*)d_in[13];
  const float* b_rp   = (const float*)d_in[14];
  float* out = (float*)d_out;

  char* ws = (char*)d_ws;
  _Float16* featA = (_Float16*)(ws);                      // 33,554,432 B NHWC
  _Float16* featB = (_Float16*)(ws + 33554432UL);         // 33,554,432 B NHWC
  float*    nv    = (float*)(ws + 67108864UL);            // 16,777,216 B planar
  float*    nv2   = (float*)(ws + 83886080UL);            // 16,777,216 B planar
  char* wp = ws + 100663296UL;
  _Float16* Wspa = (_Float16*)(wp);                       //   442,368 B
  _Float16* Wang = (_Float16*)(wp + 442368UL);            //   786,432 B
  _Float16* Wsyn = (_Float16*)(wp + 1228800UL);           //   294,912 B
  _Float16* Wr2  = (_Float16*)(wp + 1523712UL);           //    73,728 B
  _Float16* Wrp  = (_Float16*)(wp + 1597440UL);           //   294,912 B
  _Float16* r1buf = featA;   // reuse after syn
  _Float16* r2buf = featB;

  const unsigned long S1  = 64UL * HW;       // per-image NHWC stride (halves)
  const unsigned long S4  = 4UL * 64 * HW;   // per-batch NHWC stride (halves)
  const size_t lds64  = 3UL * 132 * 64 * sizeof(_Float16);   //  50,688 B
  const size_t lds256 = 3UL * 132 * 256 * sizeof(_Float16);  // 202,752 B

  // Repack weights to f16 GEMM layouts.
  repack_spa<<<(6 * 64 * 576 + 255) / 256, 256, 0, stream>>>(w_spa, Wspa,
                                                             6 * 64 * 576);
  repack_ang<<<(6 * 256 * 256 + 255) / 256, 256, 0, stream>>>(w_ang, Wang);
  repack_synrp<<<(64 * 2304 + 255) / 256, 256, 0, stream>>>(w_syn, Wsyn, 60);
  repack_synrp<<<(64 * 2304 + 255) / 256, 256, 0, stream>>>(w_rp, Wrp, 60);
  repack_r2<<<(64 * 576 + 255) / 256, 256, 0, stream>>>(w_r2, Wr2);

  // feat = lrelu(conv4d(lf_input, w_feat))  -> featA NHWC
  feat_conv<<<16 * HW / 256, 256, 0, stream>>>(lf, w_feat, b_feat, featA);

  // 6 SAS blocks: spatial WMMA conv then angular WMMA GEMM.
  for (int l = 0; l < 6; ++l) {
    conv3x3_wmma<64><<<16 * 128, 256, lds64, stream>>>(
        featA, S1, Wspa + (unsigned long)l * 64 * 576, b_spa + l * 64, 64, 1,
        featB, nullptr, nullptr, 0UL);
    gemm256_wmma<<<4 * 512, 256, 0, stream>>>(
        featB, Wang + (unsigned long)l * 256 * 256, b_ang + l * 64, featA);
  }

  // new_views = conv4d(feat, w_syn): CIN=256 per batch, planar f32 out.
  conv3x3_wmma<256><<<4 * 128, 256, lds256, stream>>>(
      featA, S4, Wsyn, b_syn, 60, 0, nullptr, nv, nullptr, S1);

  // lf_recon -> first output half.
  compose_views<<<4 * 64 * HW / 256, 256, 0, stream>>>(lf, nv, out);

  // r1 (direct), NHWC blocked layout [(b,uv)][pix][64].
  r1_conv<<<4 * 16 * HW / 256, 256, 0, stream>>>(lf, nv, w_r1, b_r1, r1buf);

  // r2: 16 "images" x CIN=64 standard conv.
  conv3x3_wmma<64><<<16 * 128, 256, lds64, stream>>>(
      r1buf, S1, Wr2, b_r2, 64, 1, r2buf, nullptr, nullptr, 0UL);

  // residual + new_views fused: rp conv (CIN=256) with f32 residual add.
  conv3x3_wmma<256><<<4 * 128, 256, lds256, stream>>>(
      r2buf, S4, Wrp, b_rp, 60, 0, nullptr, nv2, nv, S1);

  // refined recon -> second output half.
  compose_views<<<4 * 64 * HW / 256, 256, 0, stream>>>(lf, nv2,
                                                       out + 4UL * 64 * HW);
}
